// MyLSTMModel_68796786147466
// MI455X (gfx1250) — matrix operand order repaired and checked by reading, not compiled, not run on hardware
//
#include <hip/hip_runtime.h>

#define BB 8
#define TT 32
#define HH 64
#define WW 64
#define CC 3
#define FF 24
#define UU 128
#define NF (BB*TT)

typedef float  f32x8   __attribute__((ext_vector_type(8)));
typedef __bf16 bf16x16 __attribute__((ext_vector_type(16)));

union BFrag { bf16x16 v; unsigned int u[8]; uint4 q[2]; };

__device__ __forceinline__ unsigned short f2bf(float f){
  unsigned int u = __builtin_bit_cast(unsigned int, f);
  u += 0x7FFFu + ((u >> 16) & 1u);               // round-to-nearest-even
  return (unsigned short)(u >> 16);
}
__device__ __forceinline__ unsigned int pack2bf(float a, float b){
  return (unsigned int)f2bf(a) | ((unsigned int)f2bf(b) << 16);
}

// ---------------------------------------------------------------------------
// Prep: swizzle conv2 weights into the exact B-fragment register image.
// Consumption order (u32): ((s*2+nt)*32 + lane)*8 + g
// B k order used here (contiguous, 32x16): lanes0-15 N=lane K=0..15 (VGPR g -> K=2g,2g+1);
//                           lanes16-31 N=lane-16 K=16..31.
// K index of conv2: k = (ky*3+kx)*24 + ci, 216..223 zero pad; N 24..31 zero.
// ---------------------------------------------------------------------------
__global__ void prep_conv2_w(const float* __restrict__ w2, unsigned int* __restrict__ wB){
  int i = blockIdx.x * blockDim.x + threadIdx.x;
  if (i >= 7*2*32*8) return;
  int g    = i & 7;
  int lane = (i >> 3) & 31;
  int nt   = (i >> 8) & 1;
  int s    = i >> 9;
  int laneHi = lane >> 4, nl = lane & 15;
  int n  = nt*16 + nl;
  int k0 = 32*s + 2*g + (laneHi ? 16 : 0);
  float v0 = (k0     < 216 && n < FF) ? w2[k0*FF + n]     : 0.f;
  float v1 = (k0 + 1 < 216 && n < FF) ? w2[(k0+1)*FF + n] : 0.f;
  wB[i] = pack2bf(v0, v1);
}

// LSTM weights: consumption order (u32): (((gate*5+s)*8+nt)*32 + lane)*8 + g
// K = z index: 0..151 valid (z=[xt(24), h(128)]), 152..159 zero pad.
__global__ void prep_lstm_w(const float* __restrict__ Wf, const float* __restrict__ Wi,
                            const float* __restrict__ Wc, const float* __restrict__ Wo,
                            unsigned int* __restrict__ wG){
  int i = blockIdx.x * blockDim.x + threadIdx.x;
  if (i >= 4*5*8*32*8) return;
  int g    = i & 7;
  int lane = (i >> 3) & 31;
  int nt   = (i >> 8) & 7;
  int gs   = i >> 11;           // gate*5 + s
  int gate = gs / 5, s = gs - gate*5;
  const float* W = gate==0 ? Wf : gate==1 ? Wi : gate==2 ? Wc : Wo;
  int laneHi = lane >> 4, nl = lane & 15;
  int n  = nt*16 + nl;
  int k0 = 32*s + 2*g + (laneHi ? 16 : 0);
  float v0 = (k0     < 152) ? W[k0*UU + n]     : 0.f;
  float v1 = (k0 + 1 < 152) ? W[(k0+1)*UU + n] : 0.f;
  wG[i] = pack2bf(v0, v1);
}

__global__ void zero_f32(float* __restrict__ p, int n){
  int i = blockIdx.x * blockDim.x + threadIdx.x;
  if (i < n) p[i] = 0.f;
}

// ---------------------------------------------------------------------------
// Fused conv1 (VALU, fp32) + conv2 (WMMA bf16) + global-average-pool partial.
// One block = one frame x 2 output rows (y0, y0+1). 8 waves:
//   wave w: output row y0 + (w/4), pixels x = (w%4)*16 .. +15 (16-pixel M tile)
// LDS: x slab (6 rows, halo-padded cols, fp32), conv1 out slab (4 rows x 66
// padded cols x 24ch, bf16) = implicit-GEMM A source.
// ---------------------------------------------------------------------------
__global__ __launch_bounds__(256) void conv_fused(
    const float* __restrict__ x, const float* __restrict__ w1, const float* __restrict__ b1,
    const uint4* __restrict__ wB4, float* __restrict__ featsum){
  __shared__ float xs[6*68*3];              // x slab, cols shifted +2, zero borders
  __shared__ float w1s[27*24 + 24];         // conv1 weights + bias
  __shared__ unsigned short slab[4*66*24];  // conv1 output, bf16, 1px halo cols

  const int tid = threadIdx.x;
  const int f   = blockIdx.x >> 5;
  const int rb  = blockIdx.x & 31;
  const int y0  = rb * 2;

  for (int i = tid; i < 6*68*3; i += 256) xs[i] = 0.f;
  for (int i = tid; i < 672;    i += 256) w1s[i] = (i < 648) ? w1[i] : b1[i-648];
  __syncthreads();

  for (int i = tid; i < 6*64*3; i += 256){
    int row = i / 192, rem = i - row*192;
    int col = rem / 3, c = rem - col*3;
    int yy = y0 - 2 + row;
    float v = 0.f;
    if (yy >= 0 && yy < 64) v = x[((f*64 + yy)*64 + col)*3 + c];
    xs[row*204 + (col+2)*3 + c] = v;
  }
  __syncthreads();

  // conv1 into slab (rows y0-1 .. y0+2 of conv1 output, zero outside / halo)
  for (int p = tid; p < 264; p += 256){
    int r = p / 66, xp = p - r*66;
    int base = p * 24;
    int yy = y0 - 1 + r;
    if (xp == 0 || xp == 65 || yy < 0 || yy > 63){
      #pragma unroll
      for (int ch = 0; ch < 24; ch += 2) *(unsigned int*)&slab[base+ch] = 0u;
    } else {
      float acc[24];
      #pragma unroll
      for (int ch = 0; ch < 24; ++ch) acc[ch] = w1s[648+ch];
      #pragma unroll
      for (int ky = 0; ky < 3; ++ky){
        #pragma unroll
        for (int kx = 0; kx < 3; ++kx){
          const float* xp3 = &xs[(r+ky)*204 + (xp+kx)*3];
          #pragma unroll
          for (int ci = 0; ci < 3; ++ci){
            float xv = xp3[ci];
            const float* wr = &w1s[((ky*3+kx)*3 + ci)*24];
            #pragma unroll
            for (int ch = 0; ch < 24; ++ch) acc[ch] += xv * wr[ch];
          }
        }
      }
      #pragma unroll
      for (int ch = 0; ch < 24; ch += 2)
        *(unsigned int*)&slab[base+ch] = pack2bf(acc[ch], acc[ch+1]);
    }
  }
  __syncthreads();

  // conv2 implicit GEMM: M=16 pixels, N=2x16 (24ch + pad), K=7x32 (216 + pad)
  const int wv = tid >> 5, lane = tid & 31;
  const int laneHi = lane >> 4, mrow = lane & 15;
  const int waveRow = wv >> 2;
  const int xw0 = (wv & 3) * 16;
  f32x8 c0{}, c1{};
  #pragma unroll
  for (int s = 0; s < 7; ++s){
    BFrag a;
    #pragma unroll
    for (int g = 0; g < 8; ++g){
      // A k order used here (contiguous, as B): lanes0-15 K=0..15 (VGPR g -> K=2g,2g+1), lanes16-31 K=16..31
      int klocal = 2*g + (laneHi ? 16 : 0);
      int kk = 32*s + klocal;
      unsigned int v = 0u;
      if (kk < 216){
        int ky = kk / 72; int rem = kk - ky*72;
        int kx = rem / 24; int ci = rem - kx*24;
        v = *(const unsigned int*)&slab[((ky + waveRow)*66 + (xw0 + mrow + kx))*24 + ci];
      }
      a.u[g] = v;
    }
    BFrag b0, b1f;
    const uint4* p0 = wB4 + ((s*2 + 0)*32 + lane)*2;
    const uint4* p1 = wB4 + ((s*2 + 1)*32 + lane)*2;
    b0.q[0]  = p0[0]; b0.q[1]  = p0[1];
    b1f.q[0] = p1[0]; b1f.q[1] = p1[1];
    c0 = __builtin_amdgcn_wmma_f32_16x16x32_bf16(false, a.v, false, b0.v,  (short)0, c0, false, false);
    c1 = __builtin_amdgcn_wmma_f32_16x16x32_bf16(false, a.v, false, b1f.v, (short)0, c1, false, false);
  }

  // GAP partials: sum over the 16 M pixels of this tile, per output channel
  float s0 = 0.f, s1 = 0.f;
  #pragma unroll
  for (int r = 0; r < 8; ++r){ s0 += c0[r]; s1 += c1[r]; }
  s0 += __shfl_xor(s0, 16, 32);   // combine M 0-7 with M 8-15 halves
  s1 += __shfl_xor(s1, 16, 32);
  if (laneHi == 0){
    atomicAdd(&featsum[f*24 + mrow], s0);          // n = mrow < 16
    int n1 = 16 + mrow;
    if (n1 < 24) atomicAdd(&featsum[f*24 + n1], s1);
  }
}

__global__ void finalize_feat(const float* __restrict__ featsum,
                              const float* __restrict__ b2, float* __restrict__ feat){
  int i = blockIdx.x * blockDim.x + threadIdx.x;
  if (i >= NF*24) return;
  int ch = i % 24;
  feat[i] = featsum[i] * (1.f/4096.f) + b2[ch];
}

// ---------------------------------------------------------------------------
// LSTM: one 512-thread block (16 waves). Per timestep:
//  z = [xt(24), h(128)] -> 16x160 bf16 LDS tile (rows 8..15, K 152..159 zero)
//  wave w: gate = w/4, N-tiles 2*(w%4), 2*(w%4)+1 -> 5 k-step WMMA each
//  pointwise gates + state update + output dense fused.
// ---------------------------------------------------------------------------
__global__ __launch_bounds__(512) void lstm_kernel(
    const float* __restrict__ feat, const uint4* __restrict__ wG4,
    const float* __restrict__ bfv, const float* __restrict__ biv,
    const float* __restrict__ bcv, const float* __restrict__ bov,
    const float* __restrict__ ow, const float* __restrict__ ob,
    float* __restrict__ out){
  __shared__ unsigned short zbuf[16*160];
  __shared__ float hbuf[8*128];
  __shared__ float cbuf[8*128];
  __shared__ float gbuf[4*8*128];
  __shared__ float outacc[8];
  const int tid = threadIdx.x;

  for (int i = tid; i < 16*160; i += 512) zbuf[i] = 0;
  for (int i = tid; i < 1024;   i += 512){ hbuf[i] = 0.f; cbuf[i] = 0.f; }
  __syncthreads();

  const int wv = tid >> 5, lane = tid & 31, laneHi = lane >> 4, nl = lane & 15;
  const int gate = wv >> 2, ntb = (wv & 3) * 2;
  const float* bg = gate==0 ? bfv : gate==1 ? biv : gate==2 ? bcv : bov;
  const float bias0 = bg[(ntb+0)*16 + nl];
  const float bias1 = bg[(ntb+1)*16 + nl];

  for (int t = 0; t < TT; ++t){
    if (tid < 8) outacc[tid] = 0.f;
    for (int i = tid; i < 8*152; i += 512){
      int b = i / 152, k = i - b*152;
      float v = (k < 24) ? feat[(b*TT + t)*24 + k] : hbuf[b*128 + (k - 24)];
      zbuf[b*160 + k] = f2bf(v);
    }
    __syncthreads();

    f32x8 a0, a1;
    #pragma unroll
    for (int r = 0; r < 8; ++r){ a0[r] = bias0; a1[r] = bias1; }
    #pragma unroll
    for (int s = 0; s < 5; ++s){
      BFrag a;
      #pragma unroll
      for (int g = 0; g < 8; ++g){
        int klocal = 2*g + (laneHi ? 16 : 0);
        a.u[g] = *(const unsigned int*)&zbuf[nl*160 + 32*s + klocal];
      }
      BFrag b0, b1;
      const uint4* p0 = wG4 + (((gate*5 + s)*8 + ntb + 0)*32 + lane)*2;
      const uint4* p1 = wG4 + (((gate*5 + s)*8 + ntb + 1)*32 + lane)*2;
      b0.q[0] = p0[0]; b0.q[1] = p0[1];
      b1.q[0] = p1[0]; b1.q[1] = p1[1];
      a0 = __builtin_amdgcn_wmma_f32_16x16x32_bf16(false, a.v, false, b0.v, (short)0, a0, false, false);
      a1 = __builtin_amdgcn_wmma_f32_16x16x32_bf16(false, a.v, false, b1.v, (short)0, a1, false, false);
    }
    if (laneHi == 0){    // valid rows M=0..7 (batch) live in lanes 0-15
      #pragma unroll
      for (int r = 0; r < 8; ++r){
        gbuf[(gate*8 + r)*128 + (ntb+0)*16 + lane] = a0[r];
        gbuf[(gate*8 + r)*128 + (ntb+1)*16 + lane] = a1[r];
      }
    }
    __syncthreads();

    for (int i = tid; i < 1024; i += 512){
      int b = i >> 7, u = i & 127;
      float fg = 1.f / (1.f + __expf(-gbuf[(0*8 + b)*128 + u]));
      float ig = 1.f / (1.f + __expf(-gbuf[(1*8 + b)*128 + u]));
      float gg = tanhf(gbuf[(2*8 + b)*128 + u]);
      float og = 1.f / (1.f + __expf(-gbuf[(3*8 + b)*128 + u]));
      float cn = cbuf[i]*fg + ig*gg;
      float hn = tanhf(cn)*og;
      cbuf[i] = cn; hbuf[i] = hn;
      atomicAdd(&outacc[b], hn * ow[u]);
    }
    __syncthreads();
    if (tid < 8) out[tid*TT + t] = outacc[tid] + ob[0];
    __syncthreads();
  }
}

// ---------------------------------------------------------------------------
extern "C" void kernel_launch(void* const* d_in, const int* in_sizes, int n_in,
                              void* d_out, int out_size, void* d_ws, size_t ws_size,
                              hipStream_t stream){
  (void)in_sizes; (void)n_in; (void)out_size; (void)ws_size;
  const float* x   = (const float*)d_in[0];
  const float* w1  = (const float*)d_in[1];
  const float* b1  = (const float*)d_in[2];
  const float* w2  = (const float*)d_in[3];
  const float* b2  = (const float*)d_in[4];
  const float* Wf  = (const float*)d_in[5];
  const float* bfv = (const float*)d_in[6];
  const float* Wi  = (const float*)d_in[7];
  const float* biv = (const float*)d_in[8];
  const float* Wc  = (const float*)d_in[9];
  const float* bcv = (const float*)d_in[10];
  const float* Wo  = (const float*)d_in[11];
  const float* bov = (const float*)d_in[12];
  const float* ow  = (const float*)d_in[13];
  const float* ob  = (const float*)d_in[14];
  float* out = (float*)d_out;

  char* ws = (char*)d_ws;
  unsigned int* wB = (unsigned int*)(ws);             // 14336 B (conv2 B-frags)
  unsigned int* wG = (unsigned int*)(ws + 16384);     // 163840 B (LSTM B-frags)
  float* featsum   = (float*)(ws + 180224);           // 24576 B
  float* feat      = (float*)(ws + 204800);           // 24576 B

  hipLaunchKernelGGL(prep_conv2_w, dim3(14),    dim3(256), 0, stream, w2, wB);
  hipLaunchKernelGGL(prep_lstm_w,  dim3(160),   dim3(256), 0, stream, Wf, Wi, Wc, Wo, wG);
  hipLaunchKernelGGL(zero_f32,     dim3(24),    dim3(256), 0, stream, featsum, NF*24);
  hipLaunchKernelGGL(conv_fused,   dim3(NF*32), dim3(256), 0, stream,
                     x, w1, b1, (const uint4*)wB, featsum);
  hipLaunchKernelGGL(finalize_feat, dim3(24),   dim3(256), 0, stream, featsum, b2, feat);
  hipLaunchKernelGGL(lstm_kernel,  dim3(1),     dim3(512), 0, stream,
                     feat, (const uint4*)wG, bfv, biv, bcv, bov, ow, ob, out);
}
